// GLayer_60816736912088
// MI455X (gfx1250) — matrix operand order repaired and checked by reading, not compiled, not run on hardware
//
#include <hip/hip_runtime.h>
#include <hip/hip_bf16.h>

#define DIM     2048
#define RANK    64
#define BATCH   8192
#define GHID    512
#define BASE_DT 0.1f

typedef __attribute__((ext_vector_type(16))) __bf16 v16bf;
typedef __attribute__((ext_vector_type(8)))  __bf16 v8bf;
typedef __attribute__((ext_vector_type(8)))  float  v8f;

union V16u { v16bf v; v8bf h[2]; };

static __device__ inline v8f wmma_bf16(v16bf a, v16bf b, v8f c) {
  // D(16x16,f32) = A(16x32,bf16) * B(32x16,bf16) + C
  return __builtin_amdgcn_wmma_f32_16x16x32_bf16(false, a, false, b, (short)0, c,
                                                 false, false);
}

// ---- Fragment loaders -------------------------------------------------------
// B fragment (packed by k_pack_b): frag = kt*(N/16)+nt; per lane 16 contiguous
// bf16 (32B) -> lanes of a wave cover 1KB contiguous => fully coalesced b128s.
static __device__ inline v16bf load_b_packed(const __bf16* __restrict__ p,
                                             int frag, int lane) {
  const __bf16* q = p + ((size_t)frag * 32 + lane) * 16;
  V16u u;
  u.h[0] = *(const v8bf*)(q);
  u.h[1] = *(const v8bf*)(q + 8);
  return u.v;
}

// A fragment from row-major bf16 (LDS tile or global):
// lane: M = lane&15, half = lane>>4; K = kbase+half*16+[0..16)
// (contiguous, matches k_pack_b B layout)
static __device__ inline v16bf load_a_rm(const __bf16* __restrict__ base,
                                         int stride, int kbase, int lane) {
  const int m = lane & 15, half = lane >> 4;
  const __bf16* q = base + (size_t)m * stride + kbase + half * 16;
  V16u u;
  u.h[0] = *(const v8bf*)(q);
  u.h[1] = *(const v8bf*)(q + 8);
  return u.v;
}

// Stage a 16 x DIM f32 tile from global into LDS as bf16 (ds_store_b64).
static __device__ inline void stage_tile16(const float* __restrict__ src,
                                           __bf16* dst, int row0) {
  const int nf4 = 16 * DIM / 4;                 // float4 count
  for (int i = threadIdx.x; i < nf4; i += blockDim.x) {
    const int r  = i >> 9;                      // DIM/4 = 512 float4 per row
    const int c4 = i & 511;
    const float4 f = ((const float4*)(src + (size_t)(row0 + r) * DIM))[c4];
    union { __bf16 b[4]; unsigned long long u; } t;
    t.b[0] = (__bf16)f.x; t.b[1] = (__bf16)f.y;
    t.b[2] = (__bf16)f.z; t.b[3] = (__bf16)f.w;
    *(unsigned long long*)(dst + (size_t)r * DIM + c4 * 4) = t.u;
  }
}

// ---- Weight packing: f32 [K,N] row-major -> bf16 B-fragment order -----------
__global__ void k_pack_b(const float* __restrict__ W, __bf16* __restrict__ dst,
                         int N) {
  const int ntiles = N >> 4;
  const int frag = blockIdx.x;
  const int kt = frag / ntiles, nt = frag - kt * ntiles;
  const int lane = threadIdx.x;
  const int n  = nt * 16 + (lane & 15);
  const int kb = kt * 32 + ((lane >> 4) << 4);
  __bf16* out = dst + ((size_t)frag * 32 + lane) * 16;
#pragma unroll
  for (int e = 0; e < 16; ++e)
    out[e] = (__bf16)W[(size_t)(kb + e) * N + n];
}

// ---- Gate: dt = 0.1*sigmoid(tanh(x@gw1+gb1)@gw2 + gb2) ----------------------
__global__ void __launch_bounds__(256) k_gate(
    const float* __restrict__ x, const __bf16* __restrict__ gw1p,
    const float* __restrict__ gb1, const float* __restrict__ gw2,
    const float* __restrict__ gb2, float* __restrict__ dt_out) {
  extern __shared__ __bf16 tile[];              // 16*DIM bf16 = 64KB
  __shared__ float acc16[16];
  const int row0 = blockIdx.x * 16;
  if (threadIdx.x < 16) acc16[threadIdx.x] = 0.f;
  stage_tile16(x, tile, row0);
  __syncthreads();

  const int wave = threadIdx.x >> 5, lane = threadIdx.x & 31, half = lane >> 4;
  const int NT = GHID / 16;                     // 32 N-tiles over 8 waves
  float gacc[8];
#pragma unroll
  for (int r = 0; r < 8; ++r) gacc[r] = 0.f;

  for (int nt = wave; nt < NT; nt += 8) {
    v8f c = {};
    for (int kt = 0; kt < DIM / 32; ++kt) {
      v16bf a = load_a_rm(tile, DIM, kt * 32, lane);
      v16bf b = load_b_packed(gw1p, kt * NT + nt, lane);
      __builtin_prefetch(gw1p + ((size_t)((kt + 1) * NT + nt) * 32 + lane) * 16,
                         0, 1);
      c = wmma_bf16(a, b, c);
    }
    const int n = nt * 16 + (lane & 15);
    const float b1 = gb1[n], w2 = gw2[n];
#pragma unroll
    for (int r = 0; r < 8; ++r)
      gacc[r] += tanhf(c[r] + b1) * w2;         // C layout: M = half*8+r, N = n
  }
  // reduce over the 16 N-lanes of each half, then across waves via LDS atomics
#pragma unroll
  for (int r = 0; r < 8; ++r) {
    float s = gacc[r];
    s += __shfl_xor(s, 1, 32);
    s += __shfl_xor(s, 2, 32);
    s += __shfl_xor(s, 4, 32);
    s += __shfl_xor(s, 8, 32);
    if ((lane & 15) == 0) atomicAdd(&acc16[half * 8 + r], s);
  }
  __syncthreads();
  if (threadIdx.x < 16) {
    const float z = acc16[threadIdx.x] + gb2[0];
    dt_out[row0 + threadIdx.x] = BASE_DT / (1.f + __expf(-z));
  }
}

// ---- Christoffel projections: p = tanh(x@Wx) * (v@U) * (v@V)  [B,RANK] bf16 -
__global__ void __launch_bounds__(128) k_proj(
    const float* __restrict__ xin, const float* __restrict__ vin,
    const __bf16* __restrict__ Wxp, const __bf16* __restrict__ Up,
    const __bf16* __restrict__ Vp, __bf16* __restrict__ pb) {
  extern __shared__ __bf16 tile[];              // 64KB, reused for x then v
  const int row0 = blockIdx.x * 16;
  const int wave = threadIdx.x >> 5, lane = threadIdx.x & 31, half = lane >> 4;
  const int nt = wave;                          // RANK/16 = 4 waves
  v8f cx = {}, cu = {}, cv = {};

  stage_tile16(xin, tile, row0);
  __syncthreads();
  for (int kt = 0; kt < DIM / 32; ++kt) {
    v16bf a = load_a_rm(tile, DIM, kt * 32, lane);
    cx = wmma_bf16(a, load_b_packed(Wxp, kt * 4 + nt, lane), cx);
  }
  __syncthreads();
  stage_tile16(vin, tile, row0);
  __syncthreads();
  for (int kt = 0; kt < DIM / 32; ++kt) {
    v16bf a = load_a_rm(tile, DIM, kt * 32, lane);
    cu = wmma_bf16(a, load_b_packed(Up, kt * 4 + nt, lane), cu);
    cv = wmma_bf16(a, load_b_packed(Vp, kt * 4 + nt, lane), cv);
  }
  const int n = nt * 16 + (lane & 15);
#pragma unroll
  for (int r = 0; r < 8; ++r) {
    const float p = tanhf(cx[r]) * cu[r] * cv[r];
    pb[(size_t)(row0 + half * 8 + r) * RANK + n] = (__bf16)p;
  }
}

// ---- Stage 1: Gamma1 = p1@Wo fused Euler predictor -> x_e, v_e (in d_out) ---
__global__ void __launch_bounds__(256) k_euler(
    const float* __restrict__ x, const float* __restrict__ v,
    const float* __restrict__ force, const __bf16* __restrict__ pb,
    const __bf16* __restrict__ Wop, const float* __restrict__ dt,
    float* __restrict__ xe, float* __restrict__ ve) {
  const int row0 = blockIdx.x * 16;
  const int wave = threadIdx.x >> 5, lane = threadIdx.x & 31, half = lane >> 4;
  const v16bf a0 = load_a_rm(pb + (size_t)row0 * RANK, RANK, 0, lane);
  const v16bf a1 = load_a_rm(pb + (size_t)row0 * RANK, RANK, 32, lane);
  float dtm[8];
#pragma unroll
  for (int r = 0; r < 8; ++r) dtm[r] = dt[row0 + half * 8 + r];
  const int NT = DIM / 16;                      // 128 N-tiles over 8 waves
  for (int nt = wave; nt < NT; nt += 8) {
    v8f c = {};
    c = wmma_bf16(a0, load_b_packed(Wop, nt, lane), c);
    c = wmma_bf16(a1, load_b_packed(Wop, NT + nt, lane), c);
    const int ncol = nt * 16 + (lane & 15);
#pragma unroll
    for (int r = 0; r < 8; ++r) {
      const size_t idx = (size_t)(row0 + half * 8 + r) * DIM + ncol;
      const float k1 = force[idx] - c[r];
      const float xv = x[idx], vv = v[idx], d = dtm[r];
      xe[idx] = xv + d * vv;                    // x_e
      ve[idx] = vv + d * k1;                    // v_e
    }
  }
}

// ---- Stage 2: Gamma2 = p2@Wo fused Heun corrector -> x_next, v_next ---------
// v_e lives in ov[]; x_next = x + 0.5dt(v+v_e); v_next = 0.5(v+v_e)+0.5dt*k2v
__global__ void __launch_bounds__(256) k_final(
    const float* __restrict__ x, const float* __restrict__ v,
    const float* __restrict__ force, const __bf16* __restrict__ pb,
    const __bf16* __restrict__ Wop, const float* __restrict__ dt,
    float* __restrict__ ox, float* __restrict__ ov) {
  const int row0 = blockIdx.x * 16;
  const int wave = threadIdx.x >> 5, lane = threadIdx.x & 31, half = lane >> 4;
  const v16bf a0 = load_a_rm(pb + (size_t)row0 * RANK, RANK, 0, lane);
  const v16bf a1 = load_a_rm(pb + (size_t)row0 * RANK, RANK, 32, lane);
  float dtm[8];
#pragma unroll
  for (int r = 0; r < 8; ++r) dtm[r] = dt[row0 + half * 8 + r];
  const int NT = DIM / 16;
  for (int nt = wave; nt < NT; nt += 8) {
    v8f c = {};
    c = wmma_bf16(a0, load_b_packed(Wop, nt, lane), c);
    c = wmma_bf16(a1, load_b_packed(Wop, NT + nt, lane), c);
    const int ncol = nt * 16 + (lane & 15);
#pragma unroll
    for (int r = 0; r < 8; ++r) {
      const size_t idx = (size_t)(row0 + half * 8 + r) * DIM + ncol;
      const float k2 = force[idx] - c[r];
      const float vev = ov[idx];                // v_e (same-thread RAW, safe)
      const float xv = x[idx], vv = v[idx], d = dtm[r];
      const float s = 0.5f * (vv + vev);
      ox[idx] = xv + d * s;                     // x_next
      ov[idx] = s + 0.5f * d * k2;              // v_next
    }
  }
}

// ---- Host-side orchestration ------------------------------------------------
extern "C" void kernel_launch(void* const* d_in, const int* in_sizes, int n_in,
                              void* d_out, int out_size, void* d_ws,
                              size_t ws_size, hipStream_t stream) {
  const float* x     = (const float*)d_in[0];
  const float* v     = (const float*)d_in[1];
  const float* force = (const float*)d_in[2];
  const float* U     = (const float*)d_in[3];
  const float* V     = (const float*)d_in[4];
  const float* Wx    = (const float*)d_in[5];
  const float* Wo    = (const float*)d_in[6];
  const float* gw1   = (const float*)d_in[7];
  const float* gb1   = (const float*)d_in[8];
  const float* gw2   = (const float*)d_in[9];
  const float* gb2   = (const float*)d_in[10];

  // workspace layout
  const size_t OFF_DT  = 0;                                   // 8192 f32
  const size_t OFF_P1  = 32768;                               // B*RANK bf16
  const size_t OFF_P2  = OFF_P1  + (size_t)BATCH * RANK * 2;
  const size_t OFF_GW1 = OFF_P2  + (size_t)BATCH * RANK * 2;
  const size_t OFF_WX  = OFF_GW1 + (size_t)DIM * GHID * 2;
  const size_t OFF_U   = OFF_WX  + (size_t)DIM * RANK * 2;
  const size_t OFF_V   = OFF_U   + (size_t)DIM * RANK * 2;
  const size_t OFF_WO  = OFF_V   + (size_t)DIM * RANK * 2;
  const size_t WS_NEED = OFF_WO  + (size_t)RANK * DIM * 2;
  if (ws_size < WS_NEED) return;

  char* ws = (char*)d_ws;
  float*  dt   = (float*)(ws + OFF_DT);
  __bf16* p1b  = (__bf16*)(ws + OFF_P1);
  __bf16* p2b  = (__bf16*)(ws + OFF_P2);
  __bf16* gw1p = (__bf16*)(ws + OFF_GW1);
  __bf16* Wxp  = (__bf16*)(ws + OFF_WX);
  __bf16* Up   = (__bf16*)(ws + OFF_U);
  __bf16* Vp   = (__bf16*)(ws + OFF_V);
  __bf16* Wop  = (__bf16*)(ws + OFF_WO);

  float* out_x = (float*)d_out;
  float* out_v = out_x + (size_t)BATCH * DIM;

  const int ROWT = BATCH / 16;                  // 512 row-tiles
  const size_t LDS = (size_t)16 * DIM * sizeof(__bf16); // 64KB tile

  // pack weights into B-fragment order (bf16)
  k_pack_b<<<dim3((DIM / 32) * (GHID / 16)), dim3(32), 0, stream>>>(gw1, gw1p, GHID);
  k_pack_b<<<dim3((DIM / 32) * (RANK / 16)), dim3(32), 0, stream>>>(Wx, Wxp, RANK);
  k_pack_b<<<dim3((DIM / 32) * (RANK / 16)), dim3(32), 0, stream>>>(U, Up, RANK);
  k_pack_b<<<dim3((DIM / 32) * (RANK / 16)), dim3(32), 0, stream>>>(V, Vp, RANK);
  k_pack_b<<<dim3((RANK / 32) * (DIM / 16)), dim3(32), 0, stream>>>(Wo, Wop, DIM);

  // gate -> dt
  k_gate<<<dim3(ROWT), dim3(256), LDS, stream>>>(x, gw1p, gb1, gw2, gb2, dt);
  // p1 = tanh(x@Wx)*(v@U)*(v@V)
  k_proj<<<dim3(ROWT), dim3(128), LDS, stream>>>(x, v, Wxp, Up, Vp, p1b);
  // Euler predictor: x_e, v_e into d_out slots
  k_euler<<<dim3(ROWT), dim3(256), 0, stream>>>(x, v, force, p1b, Wop, dt,
                                                out_x, out_v);
  // p2 from (x_e, v_e)
  k_proj<<<dim3(ROWT), dim3(128), LDS, stream>>>(out_x, out_v, Wxp, Up, Vp, p2b);
  // Heun corrector: overwrite d_out with x_next, v_next
  k_final<<<dim3(ROWT), dim3(256), 0, stream>>>(x, v, force, p2b, Wop, dt,
                                                out_x, out_v);
}
